// MultiHeadAttention_75720273429091
// MI455X (gfx1250) — matrix operand order repaired and checked by reading, not compiled, not run on hardware
//
#include <hip/hip_runtime.h>
#include <hip/hip_bf16.h>
#include <stdint.h>

// ---------------------------------------------------------------------------
// MI455X (gfx1250) MHA forward.  wave32, v_wmma_f32_16x16x32_f16 everywhere,
// TDM (tensor_load_to_lds) staging of K/V chunks, NT store of the 536MB
// probability tensor (the bandwidth roofline), coalesced b128 epilogues.
// ---------------------------------------------------------------------------

typedef __attribute__((ext_vector_type(16))) _Float16 v16h;
typedef __attribute__((ext_vector_type(8)))  _Float16 v8h;
typedef __attribute__((ext_vector_type(8)))  float    v8f;
typedef unsigned int u32x4 __attribute__((ext_vector_type(4)));
typedef int          i32x8 __attribute__((ext_vector_type(8)));
typedef int          i32x4 __attribute__((ext_vector_type(4)));

#define BB 2
#define SS 2048
#define DD 1024
#define HH 16
#define DK 64

union U16h { v16h v; v8h h[2]; _Float16 e[16]; };
union U8h  { v8h  v; _Float16 e[8]; };

static __device__ __forceinline__ v8f wmma_f16(v16h a, v16h b, v8f c) {
  return __builtin_amdgcn_wmma_f32_16x16x32_f16(false, a, false, b, (short)0, c,
                                                false, false);
}

// A-fragment (16x32 f16): lane row m = lane&15; halves 0..15 = K[2*kb8..2*kb8+15];
// kb8 = 0 / 8 by lane half.
static __device__ __forceinline__ v16h ld_a16(const _Float16* rowp, int kb8) {
  U16h u;
  u.h[0] = *(const v8h*)(rowp + 2 * kb8);
  u.h[1] = *(const v8h*)(rowp + 2 * kb8 + 8);
  return u.v;
}

static __device__ __forceinline__ v16h cvt_a32(const float* rowp, int kb8) {
  v8f x0 = *(const v8f*)(rowp + 2 * kb8);
  v8f x1 = *(const v8f*)(rowp + 2 * kb8 + 8);
  U16h u;
#pragma unroll
  for (int i = 0; i < 8; ++i) {
    u.e[i]     = (_Float16)x0[i];
    u.e[8 + i] = (_Float16)x1[i];
  }
  return u.v;
}

// B-fragment (32x16 f16) from fp32, 16 contiguous K at p.
static __device__ __forceinline__ v16h cvt_b32(const float* p) {
  v8f x0 = *(const v8f*)(p);
  v8f x1 = *(const v8f*)(p + 8);
  U16h u;
#pragma unroll
  for (int i = 0; i < 8; ++i) {
    u.e[i]     = (_Float16)x0[i];
    u.e[8 + i] = (_Float16)x1[i];
  }
  return u.v;
}

// ---------------------------------------------------------------------------
// Tensor Data Mover: 2D tile load (2-byte elements) global -> LDS.
// D# per CDNA5 ISA ch.8: group0 = {count|flags, lds_addr, gaddr_lo,
// gaddr_hi|type=2}; group1 = dims/strides; groups 2/3 zero (<=2D tensor).
// This toolchain exposes the 6-arg builtin (g0,g1,g2,g3,g4,cpol).
// ---------------------------------------------------------------------------
static __device__ __forceinline__ void tdm_load_2d(
    unsigned lds_off, const _Float16* gsrc, unsigned tile_d0, unsigned tile_d1,
    unsigned tensor_d0, unsigned tensor_d1, unsigned stride0) {
  unsigned long long ga = (unsigned long long)(size_t)gsrc;
  u32x4 g0;
  g0[0] = 1u;                                               // count=1, user
  g0[1] = lds_off;                                          // lds_addr
  g0[2] = (unsigned)(ga & 0xffffffffu);                     // gaddr[31:0]
  g0[3] = (unsigned)((ga >> 32) & 0x1ffffffu) | (2u << 30); // gaddr[56:32]|type
  i32x8 g1;
  g1[0] = (int)(1u << 16);                       // data_size=1 -> 2B elements
  g1[1] = (int)(tensor_d0 << 16);                // tensor_dim0[15:0]
  g1[2] = (int)((tensor_d0 >> 16) | (tensor_d1 << 16));
  g1[3] = (int)((tensor_d1 >> 16) | (tile_d0 << 16));
  g1[4] = (int)(tile_d1 & 0xffffu);              // tile_dim1 (tile_dim2=0)
  g1[5] = (int)stride0;                          // tensor_dim0_stride[31:0]
  g1[6] = 0;
  g1[7] = 0;
  i32x4 z4 = {0, 0, 0, 0};
  i32x8 z8 = {0, 0, 0, 0, 0, 0, 0, 0};
  __builtin_amdgcn_tensor_load_to_lds(g0, g1, z4, z4, z8, 0);
}

static __device__ __forceinline__ unsigned lds_offset_of(const void* p) {
  // generic shared pointers carry the LDS byte offset in addr[31:0]
  return (unsigned)(size_t)p;
}

// ---------------------------------------------------------------------------
// Projection: Y = X * W^T + b  (X fp32 [B*S, D], W fp32 [D, D]) -> f16.
// vt_layout==0 -> [b,h,s,dk] (LDS-bounced b128 stores)
// vt_layout==1 -> [b,h,dk,s] (per-lane packed v8h store: 8 consecutive s)
// ---------------------------------------------------------------------------
__global__ __launch_bounds__(128) void proj_f16_kernel(
    const float* __restrict__ X, const float* __restrict__ W,
    const float* __restrict__ bias, _Float16* __restrict__ dst, int vt_layout) {
  __shared__ alignas(64) _Float16 lds_t[4][16][16];

  const int lane = threadIdx.x & 31;
  const int wave = threadIdx.x >> 5;
  const int ct = blockIdx.x * 4 + wave;  // 0..63  col tile
  const int rt = blockIdx.y;             // 0..255 row tile
  const int c0 = ct * 16, r0 = rt * 16;
  const int n = lane & 15;
  const int kb8  = (lane < 16) ? 0 : 8;
  const int kb16 = (lane < 16) ? 0 : 16;
  const int half8 = (lane >> 4) << 3;

  const float bv = bias[c0 + n];
  v8f acc = {bv, bv, bv, bv, bv, bv, bv, bv};

  const float* xrow = X + (size_t)(r0 + n) * DD;  // A row m = lane&15
  const float* wrow = W + (size_t)(c0 + n) * DD;  // B col n

  for (int kc = 0; kc < DD; kc += 32) {
    v16h a = cvt_a32(xrow + kc, kb8);
    v16h b = cvt_b32(wrow + kc + kb16);
    acc = wmma_f16(a, b, acc);
  }

  if (vt_layout) {
    // lane's 8 rows are consecutive s -> one 16B store
    int row0 = r0 + half8;
    int b = row0 >> 11, s0 = row0 & 2047;
    int col = c0 + n, head = col >> 6, d = col & 63;
    U8h pk;
#pragma unroll
    for (int r = 0; r < 8; ++r) pk.e[r] = (_Float16)acc[r];
    *(v8h*)(dst + ((size_t)(b * HH + head) * DK + d) * SS + s0) = pk.v;
  } else {
    // bounce the 16x16 tile through LDS for contiguous b128 stores
#pragma unroll
    for (int r = 0; r < 8; ++r)
      lds_t[wave][r + half8][n] = (_Float16)acc[r];
    __syncthreads();
    int srow = lane & 15;
    int dh   = (lane >> 4) << 3;  // 0 or 8
    v8h vv = *(const v8h*)&lds_t[wave][srow][dh];
    int row = r0 + srow;
    int b = row >> 11, s = row & 2047;
    int head = c0 >> 6, d = (c0 & 63) + dh;
    *(v8h*)(dst + ((size_t)(b * HH + head) * SS + s) * DK + d) = vv;
  }
}

// ---------------------------------------------------------------------------
// Fused attention, one wave per 16-row query strip.
//   TDM stages each 32-key chunk of K ([32][64]) and V^T ([64][32]) into LDS.
//   pass 1: QK^T -> online row max/sum.   pass 2: recompute, p=exp(s-m)/l,
//   NT-store probs, LDS-bounce p -> A-fragment, accumulate P*V via WMMA.
// ---------------------------------------------------------------------------
__global__ __launch_bounds__(128) void attn_kernel(
    const _Float16* __restrict__ qh, const _Float16* __restrict__ kh,
    const _Float16* __restrict__ vt, const int* __restrict__ mask,
    float* __restrict__ scores_out, _Float16* __restrict__ attnout) {
  __shared__ alignas(128) _Float16 lds_k[4][32][64];  // TDM K staging
  __shared__ alignas(128) _Float16 lds_v[4][64][32];  // TDM V^T staging
  __shared__ alignas(64)  _Float16 lds_p[4][16][32];  // p C->A bounce
  __shared__ alignas(64)  _Float16 lds_o[4][16][64];  // output tile bounce

  const int lane = threadIdx.x & 31;
  const int wave = threadIdx.x >> 5;
  const int id   = blockIdx.x * 4 + wave;  // 0..4095 strips
  const int qt   = id & 127;
  const int head = (id >> 7) & 15;
  const int b    = id >> 11;

  const int n = lane & 15;
  const int kb8  = (lane < 16) ? 0 : 8;
  const int kb16 = (lane < 16) ? 0 : 16;
  const int half8 = (lane >> 4) << 3;

  const size_t bh = (size_t)(b * HH + head);
  const _Float16* qrow = qh + (bh * SS + (size_t)qt * 16 + n) * DK;
  const _Float16* khb  = kh + bh * SS * DK;   // [2048][64]
  const _Float16* vtb  = vt + bh * DK * SS;   // [64][2048]
  const int* mb = mask + (size_t)b * SS * SS;

  const unsigned ldsk_off = lds_offset_of(&lds_k[wave][0][0]);
  const unsigned ldsv_off = lds_offset_of(&lds_v[wave][0][0]);

  const v16h qA0 = ld_a16(qrow, kb8);       // Q dims 0..31
  const v16h qA1 = ld_a16(qrow + 32, kb8);  // Q dims 32..63

  float mr[8], lr[8];
#pragma unroll
  for (int r = 0; r < 8; ++r) { mr[r] = -3.0e38f; lr[r] = 0.0f; }

  // ---------- pass 1: online softmax statistics ----------
  for (int ch = 0; ch < SS / 32; ++ch) {
    asm volatile("s_wait_dscnt 0" ::: "memory");  // close WAR on lds_k
    tdm_load_2d(ldsk_off, khb + (size_t)ch * 32 * DK, DK, 32, DK, SS, DK);
    __builtin_amdgcn_s_wait_tensorcnt(0);
    asm volatile("" ::: "memory");
#pragma unroll
    for (int t = 0; t < 2; ++t) {
      const _Float16* krow = &lds_k[wave][t * 16 + n][kb16];
      v16h b0 = *(const v16h*)(krow);
      v16h b1 = *(const v16h*)(krow + 32);
      v8f acc = {0.f, 0.f, 0.f, 0.f, 0.f, 0.f, 0.f, 0.f};
      acc = wmma_f16(qA0, b0, acc);
      acc = wmma_f16(qA1, b1, acc);
      const int kcol = ch * 32 + t * 16 + n;
#pragma unroll
      for (int r = 0; r < 8; ++r) {
        int qi = qt * 16 + r + half8;
        float s = acc[r] * 0.125f;  // 1/sqrt(64)
        if (mb[(size_t)qi * SS + kcol] == 0) s = -1.0e9f;
        float tmax = s;
#pragma unroll
        for (int off = 1; off < 16; off <<= 1)
          tmax = fmaxf(tmax, __shfl_xor(tmax, off, 32));
        float mn = fmaxf(mr[r], tmax);
        float p  = __expf(s - mn);
        float ps = p;
#pragma unroll
        for (int off = 1; off < 16; off <<= 1)
          ps += __shfl_xor(ps, off, 32);
        lr[r] = lr[r] * __expf(mr[r] - mn) + ps;
        mr[r] = mn;
      }
    }
  }

  float li[8];
#pragma unroll
  for (int r = 0; r < 8; ++r) li[r] = 1.0f / lr[r];

  v8f o[4];
#pragma unroll
  for (int j = 0; j < 4; ++j) o[j] = (v8f){0.f,0.f,0.f,0.f,0.f,0.f,0.f,0.f};

  // ---------- pass 2: probabilities out + P*V ----------
  for (int ch = 0; ch < SS / 32; ++ch) {
    asm volatile("s_wait_dscnt 0" ::: "memory");  // close WAR on lds_k/lds_v
    tdm_load_2d(ldsk_off, khb + (size_t)ch * 32 * DK, DK, 32, DK, SS, DK);
    tdm_load_2d(ldsv_off, vtb + (size_t)ch * 32, 32, DK, SS, DK, SS);
    __builtin_amdgcn_s_wait_tensorcnt(0);
    asm volatile("" ::: "memory");
#pragma unroll
    for (int t = 0; t < 2; ++t) {
      const _Float16* krow = &lds_k[wave][t * 16 + n][kb16];
      v16h b0 = *(const v16h*)(krow);
      v16h b1 = *(const v16h*)(krow + 32);
      v8f acc = {0.f, 0.f, 0.f, 0.f, 0.f, 0.f, 0.f, 0.f};
      acc = wmma_f16(qA0, b0, acc);
      acc = wmma_f16(qA1, b1, acc);
      const int kcol = ch * 32 + t * 16 + n;
#pragma unroll
      for (int r = 0; r < 8; ++r) {
        int qi = qt * 16 + r + half8;
        float s = acc[r] * 0.125f;
        if (mb[(size_t)qi * SS + kcol] == 0) s = -1.0e9f;
        float p = __expf(s - mr[r]) * li[r];
        // scores_out[b, q*h + head, k] -- written once, never read -> NT
        size_t oidx = ((size_t)(b * SS + qi) * HH + head) * (size_t)SS + kcol;
        __builtin_nontemporal_store(p, scores_out + oidx);
        lds_p[wave][r + half8][t * 16 + n] = (_Float16)p;
      }
    }
    __syncthreads();
    v16h pA = ld_a16(&lds_p[wave][n][0], kb8);  // C-layout -> A-fragment
    __syncthreads();
#pragma unroll
    for (int j = 0; j < 4; ++j) {
      const _Float16* vrow = &lds_v[wave][j * 16 + n][kb16];
      v16h vb = *(const v16h*)(vrow);
      o[j] = wmma_f16(pA, vb, o[j]);
    }
  }

  // ---------- epilogue: bounce 16x64 f16 tile -> coalesced b128 stores -----
#pragma unroll
  for (int j = 0; j < 4; ++j)
#pragma unroll
    for (int r = 0; r < 8; ++r)
      lds_o[wave][r + half8][j * 16 + n] = (_Float16)o[j][r];
  __syncthreads();
  {
    int row = lane >> 1;             // 0..15
    int seg = (lane & 1) * 32;       // 0 or 32 halves (64B per lane)
    const v16h h0 = *(const v16h*)&lds_o[wave][row][seg];
    const v16h h1 = *(const v16h*)&lds_o[wave][row][seg + 16];
    _Float16* dst = attnout + (size_t)(b * SS + qt * 16 + row) * DD +
                    head * DK + seg;
    *(v16h*)(dst)      = h0;
    *(v16h*)(dst + 16) = h1;
  }
}

// ---------------------------------------------------------------------------
// Output projection: out = A * Wo^T + bo   (A f16 [B*S, D], out fp32)
// ---------------------------------------------------------------------------
__global__ __launch_bounds__(128) void outproj_kernel(
    const _Float16* __restrict__ A, const float* __restrict__ W,
    const float* __restrict__ bias, float* __restrict__ out) {
  const int lane = threadIdx.x & 31;
  const int wave = threadIdx.x >> 5;
  const int ct = blockIdx.x * 4 + wave;
  const int rt = blockIdx.y;
  const int c0 = ct * 16, r0 = rt * 16;
  const int n = lane & 15;
  const int kb8  = (lane < 16) ? 0 : 8;
  const int kb16 = (lane < 16) ? 0 : 16;
  const int half8 = (lane >> 4) << 3;

  const float bv = bias[c0 + n];
  v8f acc = {bv, bv, bv, bv, bv, bv, bv, bv};

  const _Float16* arow = A + (size_t)(r0 + n) * DD;
  const float* wrow = W + (size_t)(c0 + n) * DD;

  for (int kc = 0; kc < DD; kc += 32) {
    v16h a = ld_a16(arow + kc, kb8);
    v16h b = cvt_b32(wrow + kc + kb16);
    acc = wmma_f16(a, b, acc);
  }

#pragma unroll
  for (int r = 0; r < 8; ++r) {
    size_t idx = (size_t)(r0 + r + half8) * DD + c0 + n;
    out[idx] = acc[r];
  }
}

// ---------------------------------------------------------------------------
extern "C" void kernel_launch(void* const* d_in, const int* in_sizes, int n_in,
                              void* d_out, int out_size, void* d_ws,
                              size_t ws_size, hipStream_t stream) {
  (void)in_sizes; (void)n_in; (void)out_size; (void)ws_size;
  const float* q    = (const float*)d_in[0];
  const float* k    = (const float*)d_in[1];
  const float* v    = (const float*)d_in[2];
  const int*   mask = (const int*)d_in[3];
  const float* Wq = (const float*)d_in[4];
  const float* bq = (const float*)d_in[5];
  const float* Wk = (const float*)d_in[6];
  const float* bk = (const float*)d_in[7];
  const float* Wv = (const float*)d_in[8];
  const float* bv = (const float*)d_in[9];
  const float* Wo = (const float*)d_in[10];
  const float* bo = (const float*)d_in[11];

  float* out    = (float*)d_out;               // [B,S,D]
  float* scores = out + (size_t)BB * SS * DD;  // [B, S*h, S]

  const size_t HSZ = (size_t)BB * HH * SS * DK;  // 4M halves = 8MB
  _Float16* qh = (_Float16*)d_ws;
  _Float16* kh = qh + HSZ;
  _Float16* vt = kh + HSZ;  // transposed [b,h,dk,s]
  _Float16* at = vt + HSZ;  // attn out  [b,s,D] f16

  dim3 blk(128);
  dim3 gproj(16, 256);  // 64 col-tiles / 4 waves, 256 row-tiles

  proj_f16_kernel<<<gproj, blk, 0, stream>>>(q, Wq, bq, qh, 0);
  proj_f16_kernel<<<gproj, blk, 0, stream>>>(k, Wk, bk, kh, 0);
  proj_f16_kernel<<<gproj, blk, 0, stream>>>(v, Wv, bv, vt, 1);
  attn_kernel<<<dim3(1024), blk, 0, stream>>>(qh, kh, vt, mask, scores, at);
  outproj_kernel<<<gproj, blk, 0, stream>>>(at, Wo, bo, out);
}
